// VAE_55044300865859
// MI455X (gfx1250) — matrix operand order repaired and checked by reading, not compiled, not run on hardware
//
#include <hip/hip_runtime.h>
#include <hip/hip_bf16.h>

typedef __attribute__((ext_vector_type(16))) _Float16 v16h;
typedef __attribute__((ext_vector_type(8)))  float    v8f;

#define N_CELLS   1024
#define N_GENES   1000
#define N_LATENT  50
#define NB_ENC    16
#define NB_DEC    16
#define N_HID     16
#define ENC_K     (N_GENES * NB_ENC)   /* 16000 */
#define BN_EPS    1e-5f
#define HALF_LOG_2PI 0.9189385332046727f
#define M_TILES   (N_CELLS / 16)       /* 64 */

// ---------------- histograms ----------------
__global__ void k_hist_enc(const float* __restrict__ coord,
                           const int* __restrict__ cxg,
                           unsigned int* __restrict__ counts, int n) {
  int i = blockIdx.x * blockDim.x + threadIdx.x;
  int stride = gridDim.x * blockDim.x;
  for (; i < n; i += stride) {
    int b = (int)(coord[i] * (float)NB_ENC);
    b = b > NB_ENC - 1 ? NB_ENC - 1 : b;
    atomicAdd(&counts[(unsigned)cxg[i] * NB_ENC + b], 1u);
  }
}

__global__ void k_hist_frag(const int* __restrict__ idx,
                            unsigned int* __restrict__ fragc, int n) {
  int i = blockIdx.x * blockDim.x + threadIdx.x;
  int stride = gridDim.x * blockDim.x;
  for (; i < n; i += stride) atomicAdd(&fragc[idx[i]], 1u);
}

// ---------------- encoder GEMM: h = log1p(counts) @ W1 + b1 ----------------
__global__ void k_encoder(const unsigned int* __restrict__ counts,
                          const float* __restrict__ W1,
                          const float* __restrict__ b1,
                          float* __restrict__ h) {
  __shared__ float s[N_HID];
  int c = blockIdx.x, t = threadIdx.x;
  if (t < N_HID) s[t] = 0.f;
  __syncthreads();
  float acc[N_HID];
#pragma unroll
  for (int j = 0; j < N_HID; ++j) acc[j] = 0.f;
  const unsigned int* row = counts + (size_t)c * ENC_K;
  for (int k = t; k < ENC_K; k += blockDim.x) {
    unsigned int cv = row[k];
    if (cv) {
      float x = log1pf((float)cv);
      const float* w = W1 + (size_t)k * N_HID;
#pragma unroll
      for (int j = 0; j < N_HID; ++j) acc[j] += x * w[j];
    }
  }
#pragma unroll
  for (int j = 0; j < N_HID; ++j) atomicAdd(&s[j], acc[j]);
  __syncthreads();
  if (t < N_HID) h[c * N_HID + t] = s[t] + b1[t];
}

// ---------------- batch-norm statistics (biased var) ----------------
__global__ void k_bnstats(const float* __restrict__ h, float* __restrict__ stats) {
  __shared__ float ssum[N_HID], ssq[N_HID];
  int t = threadIdx.x;
  if (t < N_HID) { ssum[t] = 0.f; ssq[t] = 0.f; }
  __syncthreads();
  float ls[N_HID], lq[N_HID];
#pragma unroll
  for (int j = 0; j < N_HID; ++j) { ls[j] = 0.f; lq[j] = 0.f; }
  for (int c = t; c < N_CELLS; c += blockDim.x) {
#pragma unroll
    for (int j = 0; j < N_HID; ++j) {
      float v = h[c * N_HID + j];
      ls[j] += v; lq[j] += v * v;
    }
  }
#pragma unroll
  for (int j = 0; j < N_HID; ++j) { atomicAdd(&ssum[j], ls[j]); atomicAdd(&ssq[j], lq[j]); }
  __syncthreads();
  if (t < N_HID) {
    float mu = ssum[t] / (float)N_CELLS;
    float var = ssq[t] / (float)N_CELLS - mu * mu;
    stats[t] = mu; stats[N_HID + t] = var;
  }
}

// ---------------- BN affine + relu + latent heads + reparam + latent KL ----------------
__global__ void k_latent(const float* __restrict__ h, const float* __restrict__ stats,
                         const float* __restrict__ gamma, const float* __restrict__ beta,
                         const float* __restrict__ Wl, const float* __restrict__ bl,
                         const float* __restrict__ Ws, const float* __restrict__ bs,
                         const float* __restrict__ eps,
                         float* __restrict__ latent, float* __restrict__ accum) {
  __shared__ float hn[N_HID];
  __shared__ float red;
  int c = blockIdx.x, t = threadIdx.x;
  if (t == 0) red = 0.f;
  if (t < N_HID) {
    float z = (h[c * N_HID + t] - stats[t]) * rsqrtf(stats[N_HID + t] + BN_EPS);
    z = z * gamma[t] + beta[t];
    hn[t] = z > 0.f ? z : 0.f;
  }
  __syncthreads();
  float kl = 0.f;
  if (t < N_LATENT) {
    float lo = bl[t], sr = bs[t];
#pragma unroll
    for (int j = 0; j < N_HID; ++j) {
      float hv = hn[j];
      lo += hv * Wl[j * N_LATENT + t];
      sr += hv * Ws[j * N_LATENT + t];
    }
    float q = 0.1f * expf(sr);
    float e = eps[c * N_LATENT + t];
    float lat = lo + q * e;
    latent[c * N_LATENT + t] = lat;
    // logN(lat;0,1) - logN(lat;lo,q)  ==  -0.5 lat^2 + 0.5 e^2 + log q
    kl = -0.5f * lat * lat + 0.5f * e * e + logf(q);
  }
  atomicAdd(&red, kl);
  __syncthreads();
  if (t == 0) atomicAdd(&accum[0], red);
}

// ---------------- pack A fragments in WMMA 16-bit A register layout ----------------
// Apk[((mt*2 + step)*32 + lane)*16 + j], K padded 50 -> 64.
// 16-bit A 16x32: lanes 0-15 row M=lane hold K 0-7 (el 0-7) & 16-23 (el 8-15);
//                 lanes 16-31 hold K 8-15 & 24-31.
__global__ void k_packA(const float* __restrict__ latent, _Float16* __restrict__ Apk) {
  int i = blockIdx.x * blockDim.x + threadIdx.x;   // over 64*2*32*16 = 65536
  if (i >= M_TILES * 2 * 32 * 16) return;
  int j    = i & 15;
  int lane = (i >> 4) & 31;
  int s    = (i >> 9) & 1;
  int mt   = i >> 10;
  int row  = mt * 16 + (lane & 15);
  int rhalf = lane >> 4;
  int k = s * 32 + rhalf * 8 + ((j < 8) ? j : (8 + j));  // j>=8 -> +16 offset
  float v = (k < N_LATENT) ? latent[row * N_LATENT + k] : 0.f;
  Apk[i] = (_Float16)v;
}

// ---------------- pack B fragments in WMMA 16-bit B register layout ----------------
// Bpk[((g*2 + step)*32 + lane)*16 + j]
// 16-bit B 32x16: lanes 0-15 col N=lane hold K 0-7 (el 0-7) & 16-23 (el 8-15); lanes 16-31 hold K 8-15 & 24-31.
__global__ void k_packB(const float* __restrict__ lw, _Float16* __restrict__ Bpk) {
  int i = blockIdx.x * blockDim.x + threadIdx.x;   // over 1000*2*32*16 = 1,024,000
  if (i >= N_GENES * 2 * 32 * 16) return;
  int j    = i & 15;
  int lane = (i >> 4) & 31;
  int s    = (i >> 9) & 1;
  int g    = i >> 10;
  int n    = lane & 15;
  int k    = s * 32 + (lane >> 4) * 8 + ((j < 8) ? j : (8 + j));
  float v = (k < N_LATENT) ? lw[(size_t)g * (N_LATENT * NB_DEC) + k * NB_DEC + n] : 0.f;
  Bpk[i] = (_Float16)v;
}

// ---------------- decoder GEMM via WMMA: md[1024,16000] = latent @ lw ----------------
__global__ void k_decoder_wmma(const _Float16* __restrict__ Apk,
                               const _Float16* __restrict__ Bpk,
                               float* __restrict__ md) {
  int wave = threadIdx.x >> 5;
  int lane = threadIdx.x & 31;
  int tile = blockIdx.x * (blockDim.x >> 5) + wave;
  if (tile >= M_TILES * N_GENES) return;
  int mt = tile / N_GENES;       // cell tile 0..63
  int g  = tile % N_GENES;       // gene == N-tile (NB_DEC==16)

  const v16h* ap = (const v16h*)(Apk + ((size_t)(mt * 2) * 32 + lane) * 16);
  const v16h* bp = (const v16h*)(Bpk + ((size_t)(g  * 2) * 32 + lane) * 16);
  v16h a0 = ap[0],  a1 = ap[32];   // step stride = 32 lanes * 16 el / 16 el = 32 v16h
  v16h b0 = bp[0],  b1 = bp[32];

  v8f c = {};
  c = __builtin_amdgcn_wmma_f32_16x16x32_f16(false, a0, false, b0, (short)0, c, false, false);
  c = __builtin_amdgcn_wmma_f32_16x16x32_f16(false, a1, false, b1, (short)0, c, false, false);

  int col = g * NB_DEC + (lane & 15);
  int rowbase = mt * 16 + (lane >> 4) * 8;
#pragma unroll
  for (int r = 0; r < 8; ++r)
    md[(size_t)(rowbase + r) * ENC_K + col] = c[r];
}

// ---------------- per-cut mixture log-likelihood (gather served by L2) ----------------
__global__ void k_mixture(const float* __restrict__ coord,
                          const int* __restrict__ cxg,
                          const int* __restrict__ gix,
                          const float* __restrict__ md,
                          const float* __restrict__ bh,
                          float* __restrict__ accum, int n) {
  __shared__ float red;
  if (threadIdx.x == 0) red = 0.f;
  __syncthreads();
  float part = 0.f;
  int i = blockIdx.x * blockDim.x + threadIdx.x;
  int stride = gridDim.x * blockDim.x;
  for (; i < n; i += stride) {
    const float* mrow = md + (size_t)cxg[i] * NB_DEC;
    const float* brow = bh + (size_t)gix[i] * NB_DEC;
    float lg[NB_DEC], m = -1e30f;
#pragma unroll
    for (int k = 0; k < NB_DEC; ++k) { float v = brow[k] + mrow[k]; lg[k] = v; m = v > m ? v : m; }
    float s = 0.f;
#pragma unroll
    for (int k = 0; k < NB_DEC; ++k) s += expf(lg[k] - m);
    int b = (int)(coord[i] * (float)NB_DEC); b = b > NB_DEC - 1 ? NB_DEC - 1 : b;
    part += lg[b] - m - logf(s) + 2.772588722239781f;   // + log(16)
  }
  atomicAdd(&red, part);
  __syncthreads();
  if (threadIdx.x == 0) atomicAdd(&accum[1], red);
}

// ---------------- rho_delta + Poisson fragment likelihood ----------------
__global__ void k_poisson(const float* __restrict__ latent,
                          const float* __restrict__ rw,
                          const float* __restrict__ rho_bias,
                          const int* __restrict__ libsize,
                          const unsigned int* __restrict__ fragc,
                          float* __restrict__ accum) {
  __shared__ float red;
  if (threadIdx.x == 0) red = 0.f;
  __syncthreads();
  int i = blockIdx.x * blockDim.x + threadIdx.x;
  float part = 0.f;
  if (i < N_CELLS * N_GENES) {
    int c = i / N_GENES, g = i % N_GENES;
    const float* lp = latent + c * N_LATENT;
    const float* wp = rw + g * N_LATENT;
    float rd = 0.f;
#pragma unroll 10
    for (int l = 0; l < N_LATENT; ++l) rd += lp[l] * wp[l];
    float fe = rho_bias[g] * expf(rd) * (float)libsize[c];
    float fc = (float)fragc[i];
    part = fc * logf(fe) - fe - lgammaf(fc + 1.f);
  }
  atomicAdd(&red, part);
  __syncthreads();
  if (threadIdx.x == 0) atomicAdd(&accum[2], red);
}

// ---------------- KL of decoder weights (logit_weight and rho_weight, scale 0.1) ----------------
__global__ void k_wkl(const float* __restrict__ lw, const float* __restrict__ rw,
                      float* __restrict__ accum) {
  const int NLW = N_GENES * N_LATENT * NB_DEC;       // 800000
  const int NT  = NLW + N_GENES * N_LATENT;          // + 50000
  __shared__ float red;
  if (threadIdx.x == 0) red = 0.f;
  __syncthreads();
  float part = 0.f;
  int i = blockIdx.x * blockDim.x + threadIdx.x;
  int stride = gridDim.x * blockDim.x;
  const float cst = 2.302585092994046f - HALF_LOG_2PI;  // -log(0.1) - 0.5 log 2pi
  for (; i < NT; i += stride) {
    float w = (i < NLW) ? lw[i] : rw[i - NLW];
    part += -50.f * w * w + cst;
  }
  atomicAdd(&red, part);
  __syncthreads();
  if (threadIdx.x == 0) atomicAdd(&accum[3], red);
}

// ---------------- final scalar combine ----------------
__global__ void k_final(const float* __restrict__ accum, float* __restrict__ out) {
  float lik = accum[1] * 12.5f + accum[2] * (10000.f / 1024.f);
  out[0] = -lik - accum[0] * (10000.f / 1024.f) - accum[3];
}

extern "C" void kernel_launch(void* const* d_in, const int* in_sizes, int n_in,
                              void* d_out, int out_size, void* d_ws, size_t ws_size,
                              hipStream_t stream) {
  const float* coord    = (const float*)d_in[0];
  const float* eps      = (const float*)d_in[1];
  const float* enc_W1   = (const float*)d_in[2];
  const float* enc_b1   = (const float*)d_in[3];
  const float* bn_gamma = (const float*)d_in[4];
  const float* bn_beta  = (const float*)d_in[5];
  const float* W_loc    = (const float*)d_in[6];
  const float* b_loc    = (const float*)d_in[7];
  const float* W_scale  = (const float*)d_in[8];
  const float* b_scale  = (const float*)d_in[9];
  const float* lw       = (const float*)d_in[10];
  const float* rw       = (const float*)d_in[11];
  const float* bh       = (const float*)d_in[12];
  const float* rho_bias = (const float*)d_in[13];
  const int*   cut_cxg  = (const int*)d_in[14];
  const int*   cut_gix  = (const int*)d_in[15];
  const int*   frag_ix  = (const int*)d_in[16];
  const int*   libsize  = (const int*)d_in[19];
  float* out = (float*)d_out;

  const int n_cuts  = in_sizes[0];
  const int n_frags = in_sizes[16];

  // workspace layout (counts aliased with md: counts dead after k_encoder,
  // md written later -> stream order makes the alias safe)
  char* ws = (char*)d_ws;
  size_t offCounts = 0;                                   // 65,536,000 B (u32) / md f32
  size_t offFrag   = offCounts + (size_t)N_CELLS * ENC_K * 4;
  size_t offH      = offFrag   + (size_t)N_CELLS * N_GENES * 4;
  size_t offStats  = offH      + (size_t)N_CELLS * N_HID * 4;
  size_t offLat    = offStats  + 256;
  size_t offApk    = offLat    + (size_t)N_CELLS * N_LATENT * 4;
  size_t offBpk    = offApk    + (size_t)M_TILES * 2 * 32 * 16 * 2;
  size_t offAcc    = offBpk    + (size_t)N_GENES * 2 * 32 * 16 * 2;

  unsigned int* counts = (unsigned int*)(ws + offCounts);
  float*        md     = (float*)(ws + offCounts);
  unsigned int* fragc  = (unsigned int*)(ws + offFrag);
  float*        h      = (float*)(ws + offH);
  float*        stats  = (float*)(ws + offStats);
  float*        latent = (float*)(ws + offLat);
  _Float16*     Apk    = (_Float16*)(ws + offApk);
  _Float16*     Bpk    = (_Float16*)(ws + offBpk);
  float*        accum  = (float*)(ws + offAcc);

  hipMemsetAsync(counts, 0, (size_t)N_CELLS * ENC_K * 4, stream);
  hipMemsetAsync(fragc,  0, (size_t)N_CELLS * N_GENES * 4, stream);
  hipMemsetAsync(accum,  0, 256, stream);

  k_hist_enc<<<2048, 256, 0, stream>>>(coord, cut_cxg, counts, n_cuts);
  k_hist_frag<<<2048, 256, 0, stream>>>(frag_ix, fragc, n_frags);
  k_encoder<<<N_CELLS, 256, 0, stream>>>(counts, enc_W1, enc_b1, h);
  k_bnstats<<<1, 256, 0, stream>>>(h, stats);
  k_latent<<<N_CELLS, 64, 0, stream>>>(h, stats, bn_gamma, bn_beta,
                                       W_loc, b_loc, W_scale, b_scale,
                                       eps, latent, accum);
  k_packA<<<(M_TILES * 2 * 32 * 16 + 255) / 256, 256, 0, stream>>>(latent, Apk);
  k_packB<<<(N_GENES * 2 * 32 * 16 + 255) / 256, 256, 0, stream>>>(lw, Bpk);
  k_decoder_wmma<<<M_TILES * N_GENES / 8, 256, 0, stream>>>(Apk, Bpk, md);
  k_mixture<<<4096, 256, 0, stream>>>(coord, cut_cxg, cut_gix, md, bh, accum, n_cuts);
  k_poisson<<<(N_CELLS * N_GENES + 255) / 256, 256, 0, stream>>>(latent, rw, rho_bias,
                                                                 libsize, fragc, accum);
  k_wkl<<<1024, 256, 0, stream>>>(lw, rw, accum);
  k_final<<<1, 1, 0, stream>>>(accum, out);
  (void)n_in; (void)out_size; (void)ws_size;
}
